// DeformConv2d_17987323036393
// MI455X (gfx1250) — compile-verified, matrix operand order checked by reading, not run on hardware
//
#include <hip/hip_runtime.h>
#include <hip/hip_bf16.h>
#include <stdint.h>

// ---------------------------------------------------------------------------
// Problem constants (from reference): B=4, CI=256, H=W=64, CO=256, K=3,
// stride=1, pad=1, dil=1  ->  Ho=Wo=64.
// ---------------------------------------------------------------------------
#define BATCH 4
#define CIN   256
#define COUT  256
#define HH    64
#define WW    64
#define HW    (HH * WW)          // 4096
#define KD    (CIN * 9)          // 2304  (GEMM K dimension)

typedef __attribute__((ext_vector_type(16))) __bf16 v16bf;
typedef __attribute__((ext_vector_type(8)))  float  v8f;

__device__ __forceinline__ unsigned short f32_to_bf16_rne(float f) {
    union { float f; uint32_t u; } c; c.f = f;
    uint32_t u = c.u;
    uint32_t r = u + 0x7FFFu + ((u >> 16) & 1u);   // round-to-nearest-even
    return (unsigned short)(r >> 16);
}

// ---------------------------------------------------------------------------
// Kernel 1: weight f32 -> bf16 (layout [CO][ci*9+ij] already matches GEMM A)
// ---------------------------------------------------------------------------
__global__ void cvt_weight_bf16(const float* __restrict__ in,
                                unsigned short* __restrict__ out, int n) {
    int i = blockIdx.x * blockDim.x + threadIdx.x;
    if (i < n) out[i] = f32_to_bf16_rne(in[i]);
}

// ---------------------------------------------------------------------------
// Kernel 2: deformable im2col with bilinear sampling.
// One block per (b, h, w); thread = ci. Writes colT[b][hw][ci*9+ij] (bf16),
// i.e. N-major / K-minor so GEMM B tiles are contiguous along K.
// ---------------------------------------------------------------------------
__global__ void __launch_bounds__(CIN)
im2col_deform(const float* __restrict__ x,
              const float* __restrict__ off,
              unsigned short* __restrict__ colT) {
    const int pos = blockIdx.x;            // b*HW + hw
    const int b  = pos >> 12;              // / 4096
    const int hw = pos & (HW - 1);
    const int h  = hw >> 6;
    const int w  = hw & (WW - 1);
    const int ci = threadIdx.x;

    const float* __restrict__ xc = x + ((size_t)(b * CIN + ci)) * HW;
    const float* __restrict__ ob = off + ((size_t)b) * 18 * HW + hw;
    unsigned short* __restrict__ dst = colT + (size_t)pos * KD + ci * 9;

    #pragma unroll
    for (int ij = 0; ij < 9; ++ij) {
        const int i = ij / 3, j = ij % 3;
        // dy/dx channels 2*ij, 2*ij+1 — uniform across the block
        const float oy = ob[(size_t)(2 * ij) * HW];
        const float ox = ob[(size_t)(2 * ij + 1) * HW];
        // sample position in ORIGINAL (unpadded) coords; pad ring is zero so
        // zero-outside sampling is exactly equivalent to the reference.
        const float py = (float)(h + i - 1) + oy;
        const float px = (float)(w + j - 1) + ox;
        const float y0f = floorf(py), x0f = floorf(px);
        const float ly = py - y0f, lx = px - x0f;
        const int y0 = (int)y0f, x0 = (int)x0f;
        const int y1 = y0 + 1,   x1 = x0 + 1;

        const float w00 = (1.f - ly) * (1.f - lx);
        const float w01 = (1.f - ly) * lx;
        const float w10 = ly * (1.f - lx);
        const float w11 = ly * lx;

        float acc = 0.f;
        if (y0 >= 0 && y0 < HH) {
            if (x0 >= 0 && x0 < WW) acc += xc[y0 * WW + x0] * w00;
            if (x1 >= 0 && x1 < WW) acc += xc[y0 * WW + x1] * w01;
        }
        if (y1 >= 0 && y1 < HH) {
            if (x0 >= 0 && x0 < WW) acc += xc[y1 * WW + x0] * w10;
            if (x1 >= 0 && x1 < WW) acc += xc[y1 * WW + x1] * w11;
        }
        dst[ij] = f32_to_bf16_rne(acc);
    }
}

// ---------------------------------------------------------------------------
// Kernel 3: per-batch GEMM  out[256 x 4096] = A[256 x 2304] x colT ([N][K]).
// Tiles: BM=BN=128, BK=32. 4 wave32s per block (128 thr); each wave owns a
// 64x64 sub-tile = 4x4 accumulator tiles -> 16 v_wmma_f32_16x16x32_bf16 per
// K-step against 16 ds_load_b128 fragment loads (1:1 MMA:LDS ratio).
// ---------------------------------------------------------------------------
#define BM 128
#define BN 128
#define BK 32
#define KSTEPS (KD / BK)   // 72

union Frag16 { v16bf v; uint4 q[2]; };

__global__ void __launch_bounds__(128)
gemm_bf16(const unsigned short* __restrict__ wA,
          const unsigned short* __restrict__ colT,
          float* __restrict__ out) {
    // 128 rows x 32 bf16 = 8 KB each (as uint4 for 16B alignment)
    __shared__ uint4 As[BM * BK / 8];   // 512
    __shared__ uint4 Bs[BN * BK / 8];   // 512

    const int b     = blockIdx.z;
    const int mTile = blockIdx.y * BM;
    const int nTile = blockIdx.x * BN;
    const int tid   = threadIdx.x;     // 0..127
    const int lane  = tid & 31;
    const int wave  = tid >> 5;        // 0..3
    const int wm    = wave >> 1;       // 0..1 -> 64-row half
    const int wn    = wave & 1;        // 0..1 -> 64-col half

    const uint4* __restrict__ gA = (const uint4*)wA;
    const uint4* __restrict__ gB = (const uint4*)(colT + (size_t)b * HW * KD);

    v8f acc[4][4];
    #pragma unroll
    for (int i2 = 0; i2 < 4; ++i2)
        #pragma unroll
        for (int j2 = 0; j2 < 4; ++j2)
            acc[i2][j2] = (v8f){0.f, 0.f, 0.f, 0.f, 0.f, 0.f, 0.f, 0.f};

    // per-lane fragment addressing (ISA 16-bit A/B striping):
    // lanes 0-15 hold K-chunks {0..7,16..23}; lanes 16-31 hold {8..15,24..31}
    const int rl = lane & 15;
    const int kb = lane >> 4;          // 0/1 -> uint4 index kb and kb+2

    for (int kt = 0; kt < KSTEPS; ++kt) {
        const int k0 = kt * BK;
        // cooperative tile load: thread tid covers the full 32-bf16 row tid
        const size_t aIdx = ((size_t)(mTile + tid) * KD + k0) / 8;
        const size_t bIdx = ((size_t)(nTile + tid) * KD + k0) / 8;
        #pragma unroll
        for (int q = 0; q < 4; ++q) As[tid * 4 + q] = gA[aIdx + q];
        #pragma unroll
        for (int q = 0; q < 4; ++q) Bs[tid * 4 + q] = gB[bIdx + q];
        if (kt + 1 < KSTEPS) {         // gfx1250 global_prefetch of next K tile
            __builtin_prefetch(&gA[aIdx + 4], 0, 3);
            __builtin_prefetch(&gB[bIdx + 4], 0, 3);
        }
        __syncthreads();

        // keep 4 B fragments resident; stream A fragments one at a time
        Frag16 bfm[4];
        #pragma unroll
        for (int j2 = 0; j2 < 4; ++j2) {
            const int n = wn * 64 + j2 * 16 + rl;
            bfm[j2].q[0] = Bs[n * 4 + kb];
            bfm[j2].q[1] = Bs[n * 4 + kb + 2];
        }
        #pragma unroll
        for (int i2 = 0; i2 < 4; ++i2) {
            Frag16 af;
            const int m = wm * 64 + i2 * 16 + rl;
            af.q[0] = As[m * 4 + kb];
            af.q[1] = As[m * 4 + kb + 2];
            #pragma unroll
            for (int j2 = 0; j2 < 4; ++j2)
                acc[i2][j2] = __builtin_amdgcn_wmma_f32_16x16x32_bf16(
                    false, af.v, false, bfm[j2].v,
                    (short)0, acc[i2][j2], false, false);
        }

        __syncthreads();
    }

    // Epilogue: C layout — VGPR v, lanes 0-15 rows M=v, lanes 16-31 rows M=v+8
    float* __restrict__ ob = out + (size_t)b * COUT * HW;
    #pragma unroll
    for (int i2 = 0; i2 < 4; ++i2) {
        #pragma unroll
        for (int j2 = 0; j2 < 4; ++j2) {
            const int col  = nTile + wn * 64 + j2 * 16 + rl;
            const int row0 = mTile + wm * 64 + i2 * 16 + kb * 8;
            #pragma unroll
            for (int v = 0; v < 8; ++v)
                ob[(size_t)(row0 + v) * HW + col] = acc[i2][j2][v];
        }
    }
}

// ---------------------------------------------------------------------------
// Launch
// ---------------------------------------------------------------------------
extern "C" void kernel_launch(void* const* d_in, const int* in_sizes, int n_in,
                              void* d_out, int out_size, void* d_ws, size_t ws_size,
                              hipStream_t stream) {
    const float* x      = (const float*)d_in[0];   // [4,256,64,64]
    const float* offset = (const float*)d_in[1];   // [4,18,64,64]
    const float* weight = (const float*)d_in[2];   // [256,256,3,3]
    float* out = (float*)d_out;                    // [4,256,64,64]

    // workspace layout: bf16 weight [CO*KD], then colT [B*HW*KD] (bf16)
    unsigned short* wA   = (unsigned short*)d_ws;              // 1.18 MB
    unsigned short* colT = wA + (size_t)COUT * KD;             // 75.5 MB

    const int wn = COUT * KD;                                  // 589824
    cvt_weight_bf16<<<(wn + 255) / 256, 256, 0, stream>>>(weight, wA, wn);

    im2col_deform<<<BATCH * HW, CIN, 0, stream>>>(x, offset, colT);

    dim3 grid(HW / BN, COUT / BM, BATCH);                      // (32, 2, 4)
    gemm_bf16<<<grid, 128, 0, stream>>>(wA, colT, out);
}
